// ViT_74088185856637
// MI455X (gfx1250) — compile-verified, matrix operand order checked by reading, not run on hardware
//
#include <hip/hip_runtime.h>

typedef __attribute__((ext_vector_type(16))) __bf16 v16bf;
typedef __attribute__((ext_vector_type(2)))  __bf16 v2bf;
typedef __attribute__((ext_vector_type(8)))  float  v8f;
typedef unsigned short u16;
typedef unsigned int   u32;

#define AS3 __attribute__((address_space(3)))

#define C_DIM  768
#define NHEAD  12
#define HDIM   64
#define WSZ    14
#define S_REAL 196
#define SP     224          // padded window sequence (14 * 16)
#define NWIN   100          // B * 5 * 5 windows
#define BHEADS (NWIN * NHEAD)
#define HID    3072
#define TOK    (4 * 64 * 64)
#define MW     (NWIN * SP)  // 22400 padded window tokens

__device__ __forceinline__ u16 f2bf(float f) {
  u32 u = __builtin_bit_cast(u32, f);
  u32 r = u + 0x7FFFu + ((u >> 16) & 1u);
  return (u16)(r >> 16);
}
__device__ __forceinline__ float bf2f(u16 h) {
  u32 u = ((u32)h) << 16;
  return __builtin_bit_cast(float, u);
}
// two f32 -> packed bf16 in one VALU op when the HW convert is exposed
__device__ __forceinline__ u32 pack_bf16(float lo, float hi) {
#if __has_builtin(__builtin_amdgcn_cvt_pk_bf16_f32)
  v2bf r = __builtin_amdgcn_cvt_pk_bf16_f32(lo, hi);
  return __builtin_bit_cast(u32, r);
#else
  return (u32)f2bf(lo) | ((u32)f2bf(hi) << 16);
#endif
}
// branch-free erf (Abramowitz-Stegun 7.1.26, |err| <= 1.5e-7)
__device__ __forceinline__ float erf_approx(float x) {
  float ax = fabsf(x);
  float t = 1.0f / (1.0f + 0.3275911f * ax);
  float y = t * (0.254829592f +
           t * (-0.284496736f +
           t * (1.421413741f +
           t * (-1.453152027f +
           t * 1.061405429f))));
  float r = 1.0f - y * __expf(-ax * ax);
  return copysignf(r, x);
}

union Frag { u32 u[8]; uint4 q[2]; v16bf v; };

// ISA 7.12.2 16-bit A/B fragment: per lane-half the 16 elements live at
// K = half*8 + {0..7} and K = 16 + half*8 + {0..7}  ->  two 16B loads.
__device__ __forceinline__ void load_frag(Frag& f, const u16* base, int half) {
  f.q[0] = *(const uint4*)(base + half * 8);
  f.q[1] = *(const uint4*)(base + 16 + half * 8);
}

__device__ __forceinline__ v8f wmma_bf16(const Frag& a, const Frag& b, v8f c) {
  return __builtin_amdgcn_wmma_f32_16x16x32_bf16(false, a.v, false, b.v,
                                                 (short)0, c, false, false);
}

// ------------------------------------------------- weights cvt (2-wide packed)
__global__ void cvt_f32_bf16(const float* __restrict__ src, u32* __restrict__ dst, int n2) {
  int i = blockIdx.x * 256 + threadIdx.x;
  if (i < n2) {
    float2 f = ((const float2*)src)[i];
    dst[i] = pack_bf16(f.x, f.y);
  }
}

// ------------------------------------------------- LN1 + window partition (bf16)
__global__ __launch_bounds__(256)
void ln1_window_kernel(const float* __restrict__ x, const float* __restrict__ g,
                       const float* __restrict__ bt, u16* __restrict__ Tw) {
  __shared__ float red[256];
  int blk = blockIdx.x;
  int win = blk / SP, s = blk - win * SP;
  int bi = win / 25, wrem = win - bi * 25, wh = wrem / 5, ww = wrem - wh * 5;
  int sr = s / WSZ, sc = s - sr * WSZ;
  int gh = wh * WSZ + sr, gw = ww * WSZ + sc;
  int tid = threadIdx.x;
  u16* dst = Tw + (size_t)blk * C_DIM;
  bool valid = (s < S_REAL) && (gh < 64) && (gw < 64);
  if (!valid) {
    for (int c = tid; c < C_DIM; c += 256) dst[c] = 0;
    return;
  }
  const float* row = x + (((size_t)bi * 64 + gh) * 64 + gw) * C_DIM;
  float v0[3], s1 = 0.f, s2 = 0.f;
#pragma unroll
  for (int k = 0; k < 3; k++) { float v = row[tid + k * 256]; v0[k] = v; s1 += v; s2 += v * v; }
  red[tid] = s1; __syncthreads();
  for (int st = 128; st > 0; st >>= 1) { if (tid < st) red[tid] += red[tid + st]; __syncthreads(); }
  float mu = red[0] * (1.0f / C_DIM); __syncthreads();
  red[tid] = s2; __syncthreads();
  for (int st = 128; st > 0; st >>= 1) { if (tid < st) red[tid] += red[tid + st]; __syncthreads(); }
  float var = red[0] * (1.0f / C_DIM) - mu * mu;
  float rs = rsqrtf(var + 1e-5f);
#pragma unroll
  for (int k = 0; k < 3; k++) {
    int c = tid + k * 256;
    dst[c] = f2bf((v0[k] - mu) * rs * g[c] + bt[c]);
  }
}

// ------------------------------------------------------------- LN2 (x2 -> bf16)
__global__ __launch_bounds__(256)
void ln2_kernel(const float* __restrict__ x2, const float* __restrict__ g,
                const float* __restrict__ bt, u16* __restrict__ L2) {
  __shared__ float red[256];
  int tok = blockIdx.x, tid = threadIdx.x;
  const float* row = x2 + (size_t)tok * C_DIM;
  float v0[3], s1 = 0.f, s2 = 0.f;
#pragma unroll
  for (int k = 0; k < 3; k++) { float v = row[tid + k * 256]; v0[k] = v; s1 += v; s2 += v * v; }
  red[tid] = s1; __syncthreads();
  for (int st = 128; st > 0; st >>= 1) { if (tid < st) red[tid] += red[tid + st]; __syncthreads(); }
  float mu = red[0] * (1.0f / C_DIM); __syncthreads();
  red[tid] = s2; __syncthreads();
  for (int st = 128; st > 0; st >>= 1) { if (tid < st) red[tid] += red[tid + st]; __syncthreads(); }
  float var = red[0] * (1.0f / C_DIM) - mu * mu;
  float rs = rsqrtf(var + 1e-5f);
  u16* dst = L2 + (size_t)tok * C_DIM;
#pragma unroll
  for (int k = 0; k < 3; k++) {
    int c = tid + k * 256;
    dst[c] = f2bf((v0[k] - mu) * rs * g[c] + bt[c]);
  }
}

// ----------------------------------------------------- generic bf16 WMMA GEMM
#define GEMM_QKV  0
#define GEMM_PROJ 1
#define GEMM_FC1  2
#define GEMM_FC2  3

// Double-buffered pipeline: next A tile DMA'd to LDS via ASYNCcnt while the
// current tile's WMMAs run; next B tile's global loads issued before the wait,
// its LDS transpose-scatter deferred until after the WMMAs.
template <int MODE>
__global__ __launch_bounds__(256)
void gemm_bf16_kernel(const u16* __restrict__ A, const u16* __restrict__ Bm,
                      int K, int N,
                      const float* __restrict__ bias,
                      const float* __restrict__ addsrc,
                      float* __restrict__ outf,
                      u16* __restrict__ ob0, u16* __restrict__ ob1, u16* __restrict__ ob2) {
  __shared__ u16 As[2][64 * 40];    // A tile 64x32, stride 40 (16B-aligned rows)
  __shared__ u16 Bs[2][128 * 40];   // B tile transposed [n][k], stride 40
  int tid = threadIdx.x;
  int lane = tid & 31, half = lane >> 4, wid = tid >> 5;
  int l15 = lane & 15;
  int wr = wid >> 2, wc = wid & 3;
  int m0 = blockIdx.y * 64, n0 = blockIdx.x * 128;
  v8f acc[2][2] = {};
  int arow = tid >> 2, aq = tid & 3;
  int brow = tid >> 4, bcol8 = (tid & 15) * 8;

  const u16* gA  = A + (size_t)(m0 + arow) * K + aq * 8;
  const u16* gB0 = Bm + (size_t)brow * N + n0 + bcol8;
  const u16* gB1 = Bm + (size_t)(brow + 16) * N + n0 + bcol8;

  // prologue: stage tile 0
  asm volatile("global_load_async_to_lds_b128 %0, %1, off"
               :: "v"((AS3 u16*)(As[0] + arow * 40 + aq * 8)), "v"(gA) : "memory");
  {
    union { uint4 q; u16 h[8]; } t0, t1;
    t0.q = *(const uint4*)gB0;
    t1.q = *(const uint4*)gB1;
#pragma unroll
    for (int e = 0; e < 8; e++) {
      Bs[0][(bcol8 + e) * 40 + brow]      = t0.h[e];
      Bs[0][(bcol8 + e) * 40 + brow + 16] = t1.h[e];
    }
  }

  int nk = K >> 5;
  for (int kt = 0; kt < nk; kt++) {
    int cur = kt & 1, nxt = cur ^ 1;
    bool more = (kt + 1 < nk);
    union { uint4 q; u16 h[8]; } t0, t1;
    if (more) {  // issue next B-tile global loads early (latency hidden by WMMAs)
      t0.q = *(const uint4*)(gB0 + (size_t)(kt + 1) * 32 * N);
      t1.q = *(const uint4*)(gB1 + (size_t)(kt + 1) * 32 * N);
    }
    asm volatile("s_wait_asynccnt 0" ::: "memory");   // this tile's A DMA done
    __syncthreads();
    if (more) {  // kick next A-tile DMA; overlaps with this tile's WMMAs
      asm volatile("global_load_async_to_lds_b128 %0, %1, off"
                   :: "v"((AS3 u16*)(As[nxt] + arow * 40 + aq * 8)),
                      "v"(gA + (size_t)(kt + 1) * 32) : "memory");
    }
    Frag af[2], bf[2];
#pragma unroll
    for (int i = 0; i < 2; i++)
      load_frag(af[i], As[cur] + (wr * 32 + i * 16 + l15) * 40, half);
#pragma unroll
    for (int j = 0; j < 2; j++)
      load_frag(bf[j], Bs[cur] + (wc * 32 + j * 16 + l15) * 40, half);
#pragma unroll
    for (int i = 0; i < 2; i++)
#pragma unroll
      for (int j = 0; j < 2; j++)
        acc[i][j] = wmma_bf16(af[i], bf[j], acc[i][j]);
    if (more) {  // scatter next B tile (transposed) into the other buffer
#pragma unroll
      for (int e = 0; e < 8; e++) {
        Bs[nxt][(bcol8 + e) * 40 + brow]      = t0.h[e];
        Bs[nxt][(bcol8 + e) * 40 + brow + 16] = t1.h[e];
      }
    }
  }

  // epilogue: D layout -> M = j + 8*half, N = lane%16
#pragma unroll
  for (int i = 0; i < 2; i++)
#pragma unroll
    for (int jt = 0; jt < 2; jt++)
#pragma unroll
      for (int j = 0; j < 8; j++) {
        int gm = m0 + wr * 32 + i * 16 + j + half * 8;
        int gn = n0 + wc * 32 + jt * 16 + l15;
        float v = acc[i][jt][j];
        if constexpr (MODE == GEMM_QKV) {
          v += bias[gn];
          int comp = gn / C_DIM, rem = gn - comp * C_DIM;
          int head = rem >> 6, hd = rem & 63;
          int win = gm / SP, s = gm - win * SP;
          size_t bh = (size_t)(win * NHEAD + head);
          if (comp == 0)       ob0[(bh * SP + s) * HDIM + hd] = f2bf(v * 0.125f); // q * HD^-0.5
          else if (comp == 1)  ob1[(bh * SP + s) * HDIM + hd] = f2bf(v);
          else                 ob2[(bh * HDIM + hd) * SP + s] = f2bf(v);          // V transposed
        } else if constexpr (MODE == GEMM_PROJ) {
          v += bias[gn];
          int win = gm / SP, s = gm - win * SP;
          if (s < S_REAL) {
            int bi = win / 25, wrem = win - bi * 25, wh = wrem / 5, ww = wrem - wh * 5;
            int sr = s / WSZ, sc = s - sr * WSZ;
            int gh = wh * WSZ + sr, gw = ww * WSZ + sc;
            if (gh < 64 && gw < 64) {
              size_t idx = (((size_t)bi * 64 + gh) * 64 + gw) * C_DIM + gn;
              outf[idx] = addsrc[idx] + v;   // x2 = x + attn_out
            }
          }
        } else if constexpr (MODE == GEMM_FC1) {
          v += bias[gn];
          v = 0.5f * v * (1.0f + erf_approx(v * 0.70710678118f));  // exact-GELU approx
          ob0[(size_t)gm * HID + gn] = f2bf(v);
        } else { // GEMM_FC2
          v += bias[gn] + addsrc[(size_t)gm * C_DIM + gn];
          outf[(size_t)gm * C_DIM + gn] = v;
        }
      }
}

// -------------------------------------------------------------- attention core
// One block per (window, head).  Computes S^T = K*Q^T so that the D-fragment
// layout of the probabilities is exactly the B-fragment layout needed for
// out^T = V^T * P^T  (no cross-lane movement, no LDS P buffer).
__global__ __launch_bounds__(256)
void attn_kernel(const u16* __restrict__ Qb, const u16* __restrict__ Kb,
                 const u16* __restrict__ Vb, const float* __restrict__ relph,
                 const float* __restrict__ relpw, u16* __restrict__ Ct) {
  __shared__ float relH[SP * WSZ];
  __shared__ float relW[SP * WSZ];
  int bh = blockIdx.x;
  int win = bh / NHEAD, head = bh - win * NHEAD;
  int tid = threadIdx.x, lane = tid & 31, half = lane >> 4, wid = tid >> 5;
  int l15 = lane & 15;
  const u16* Qg = Qb + (size_t)bh * SP * HDIM;
  const u16* Kg = Kb + (size_t)bh * SP * HDIM;
  const u16* Vg = Vb + (size_t)bh * HDIM * SP;   // V^T: [hd][key]

  // decomposed rel-pos tables: relH[qrow][kh] = q . Rh, relW[qrow][kw] = q . Rw
  for (int e = tid; e < SP * WSZ; e += 256) {
    int qrow = e / WSZ, kk = e - qrow * WSZ;
    float sh = 0.f, sw = 0.f;
    if (qrow < S_REAL) {
      int qh = qrow / WSZ, qw = qrow - qh * WSZ;
      const u16* qr = Qg + (size_t)qrow * HDIM;
      const float* Rh = relph + (size_t)(qh - kk + WSZ - 1) * HDIM;
      const float* Rw = relpw + (size_t)(qw - kk + WSZ - 1) * HDIM;
#pragma unroll 8
      for (int c = 0; c < HDIM; c++) {
        float qv = bf2f(qr[c]);
        sh += qv * Rh[c];
        sw += qv * Rw[c];
      }
      sh *= 8.f; sw *= 8.f;   // undo the HD^-0.5 baked into Q
    }
    relH[e] = sh; relW[e] = sw;
  }
  __syncthreads();

  for (int rt = wid; rt < SP / 16; rt += 8) {
    int qrow = rt * 16 + l15;
    // Q as B-fragments (N = qrow, K = hd): 2 k-steps of 32
    Frag qf[2];
#pragma unroll
    for (int ks = 0; ks < 2; ks++)
      load_frag(qf[ks], Qg + (size_t)qrow * HDIM + ks * 32, half);

    // S^T accumulation: 14 key m-tiles
    v8f sacc[14] = {};
#pragma unroll
    for (int t = 0; t < 14; t++) {
#pragma unroll
      for (int ks = 0; ks < 2; ks++) {
        Frag kf;
        load_frag(kf, Kg + (size_t)(t * 16 + l15) * HDIM + ks * 32, half);
        sacc[t] = wmma_bf16(kf, qf[ks], sacc[t]);
      }
    }

    // rel-pos + masked softmax along keys (M dim of S^T), per fixed qrow (lane)
    float mx = -3.0e38f;
#pragma unroll
    for (int t = 0; t < 14; t++)
#pragma unroll
      for (int j = 0; j < 8; j++) {
        int key = t * 16 + half * 8 + j;
        float v;
        if (key < S_REAL)
          v = sacc[t][j] + relH[qrow * WSZ + key / WSZ] + relW[qrow * WSZ + key % WSZ];
        else
          v = -3.0e38f;
        sacc[t][j] = v;
        mx = fmaxf(mx, v);
      }
    mx = fmaxf(mx, __shfl_xor(mx, 16, 32));   // other half holds the other keys
    float sum = 0.f;
#pragma unroll
    for (int t = 0; t < 14; t++)
#pragma unroll
      for (int j = 0; j < 8; j++) {
        int key = t * 16 + half * 8 + j;
        float e = (key < S_REAL) ? __expf(sacc[t][j] - mx) : 0.f;
        sacc[t][j] = e;
        sum += e;
      }
    sum += __shfl_xor(sum, 16, 32);
    float inv = 1.0f / sum;

    // out^T = V^T @ P^T ; P^T B-fragments packed straight from sacc registers
    v8f oacc[4] = {};
#pragma unroll
    for (int ks = 0; ks < 7; ks++) {
      Frag pf;
#pragma unroll
      for (int p = 0; p < 8; p++) {
        int t = 2 * ks + (p >> 2);
        int j0 = (2 * p) & 7;
        pf.u[p] = pack_bf16(sacc[t][j0] * inv, sacc[t][j0 + 1] * inv);
      }
#pragma unroll
      for (int mt = 0; mt < 4; mt++) {
        Frag vf;
        load_frag(vf, Vg + (size_t)(mt * 16 + l15) * SP + ks * 32, half);
        oacc[mt] = wmma_bf16(vf, pf, oacc[mt]);
      }
    }

    // write context (bf16, dword-packed) at [win][qrow][head*64 + hd]
#pragma unroll
    for (int mt = 0; mt < 4; mt++) {
      u32* cw = (u32*)(Ct + ((size_t)win * SP + qrow) * C_DIM + head * HDIM
                       + mt * 16 + half * 8);
#pragma unroll
      for (int jp = 0; jp < 4; jp++)
        cw[jp] = pack_bf16(oacc[mt][2 * jp], oacc[mt][2 * jp + 1]);
    }
  }
}

// ------------------------------------------------------------------- launcher
extern "C" void kernel_launch(void* const* d_in, const int* in_sizes, int n_in,
                              void* d_out, int out_size, void* d_ws, size_t ws_size,
                              hipStream_t stream) {
  (void)in_sizes; (void)n_in; (void)out_size; (void)ws_size;
  const float* x      = (const float*)d_in[0];
  const float* ln1_g  = (const float*)d_in[1];
  const float* ln1_b  = (const float*)d_in[2];
  const float* qkv_w  = (const float*)d_in[3];
  const float* qkv_b  = (const float*)d_in[4];
  const float* proj_w = (const float*)d_in[5];
  const float* proj_b = (const float*)d_in[6];
  const float* rph    = (const float*)d_in[7];
  const float* rpw    = (const float*)d_in[8];
  const float* ln2_g  = (const float*)d_in[9];
  const float* ln2_b  = (const float*)d_in[10];
  const float* fc1_w  = (const float*)d_in[11];
  const float* fc1_b  = (const float*)d_in[12];
  const float* fc2_w  = (const float*)d_in[13];
  const float* fc2_b  = (const float*)d_in[14];
  float* out = (float*)d_out;

  char* ws = (char*)d_ws;
  size_t off = 0;
  auto alloc = [&](size_t bytes) -> void* {
    void* p = ws + off;
    off = (off + bytes + 255) & ~(size_t)255;
    return p;
  };
  const size_t TOKWIN_B = (size_t)MW * C_DIM * 2;         // 34.4 MB
  const size_t QKV1_B   = (size_t)BHEADS * SP * HDIM * 2; // 34.4 MB each

  u16* Wqkv  = (u16*)alloc((size_t)C_DIM * 3 * C_DIM * 2);
  u16* Wproj = (u16*)alloc((size_t)C_DIM * C_DIM * 2);
  u16* Wfc1  = (u16*)alloc((size_t)C_DIM * HID * 2);
  u16* Wfc2  = (u16*)alloc((size_t)HID * C_DIM * 2);
  size_t offA = off;
  u16* Tw   = (u16*)alloc(TOKWIN_B);   // LN1 window tokens (bf16)
  u16* Qb   = (u16*)alloc(QKV1_B);
  u16* Kbuf = (u16*)alloc(QKV1_B);
  u16* Vbuf = (u16*)alloc(QKV1_B);
  size_t offCt = off;
  u16* Ct   = (u16*)alloc(TOKWIN_B);   // attention context (bf16)
  float* x2 = (float*)alloc((size_t)TOK * C_DIM * 4);
  u16* Hmid = (u16*)(ws + offA);       // 100.7MB, aliases Tw+Qb+Kbuf (dead by FC1)
  u16* L2t  = (u16*)(ws + offCt);      // aliases Ct (dead by LN2)

  cvt_f32_bf16<<<(C_DIM * 3 * C_DIM / 2 + 255) / 256, 256, 0, stream>>>(
      qkv_w, (u32*)Wqkv, C_DIM * 3 * C_DIM / 2);
  cvt_f32_bf16<<<(C_DIM * C_DIM / 2 + 255) / 256, 256, 0, stream>>>(
      proj_w, (u32*)Wproj, C_DIM * C_DIM / 2);
  cvt_f32_bf16<<<(C_DIM * HID / 2 + 255) / 256, 256, 0, stream>>>(
      fc1_w, (u32*)Wfc1, C_DIM * HID / 2);
  cvt_f32_bf16<<<(HID * C_DIM / 2 + 255) / 256, 256, 0, stream>>>(
      fc2_w, (u32*)Wfc2, HID * C_DIM / 2);

  ln1_window_kernel<<<MW, 256, 0, stream>>>(x, ln1_g, ln1_b, Tw);

  gemm_bf16_kernel<GEMM_QKV><<<dim3(3 * C_DIM / 128, MW / 64), 256, 0, stream>>>(
      Tw, Wqkv, C_DIM, 3 * C_DIM, qkv_b, nullptr, nullptr, Qb, Kbuf, Vbuf);

  attn_kernel<<<BHEADS, 256, 0, stream>>>(Qb, Kbuf, Vbuf, rph, rpw, Ct);

  gemm_bf16_kernel<GEMM_PROJ><<<dim3(C_DIM / 128, MW / 64), 256, 0, stream>>>(
      Ct, Wproj, C_DIM, C_DIM, proj_b, x, x2, nullptr, nullptr, nullptr);

  ln2_kernel<<<TOK, 256, 0, stream>>>(x2, ln2_g, ln2_b, L2t);

  gemm_bf16_kernel<GEMM_FC1><<<dim3(HID / 128, TOK / 64), 256, 0, stream>>>(
      L2t, Wfc1, C_DIM, HID, fc1_b, nullptr, nullptr, Hmid, nullptr, nullptr);

  gemm_bf16_kernel<GEMM_FC2><<<dim3(C_DIM / 128, TOK / 64), 256, 0, stream>>>(
      Hmid, Wfc2, HID, C_DIM, fc2_b, x2, out, nullptr, nullptr, nullptr);
}
